// InfoNCELoss_36034775613658
// MI455X (gfx1250) — matrix operand order repaired and checked by reading, not compiled, not run on hardware
//
#include <hip/hip_runtime.h>

// InfoNCE loss for MI455X (gfx1250, wave32, WMMA).
//
// Phase 1: per-row L2 normalize x1/x2, compute positive[n] = sum_d exp(x1n*x2n),
//          pack normalized x1 rows as f16 into scratch (3.2 MB -> L2 resident).
// Phase 2: one wave per row; 100 negatives in 7 tiles of 16; each tile is
//          two v_wmma_f32_16x16x32_f16 (K=0..31, K=32..63) with
//          A = 16 gathered f16 negative rows, B = this row's vector replicated
//          across the 16 N-columns. D column 0 (any column) = the 16 dots.

typedef __attribute__((ext_vector_type(16))) _Float16 v16h;
typedef __attribute__((ext_vector_type(8)))  _Float16 v8h;
typedef __attribute__((ext_vector_type(8)))  float    v8f;

#define HW_   3136      // 56*56
#define D_    64
#define NROW_ 25088     // 8*56*56
#define KNEG_ 100
#define EPS_  1e-12f

// ---------------------------------------------------------------- phase 1 ----
__global__ void nce_normalize_kernel(const float* __restrict__ x1,
                                     const float* __restrict__ x2,
                                     _Float16* __restrict__ x1h,
                                     float* __restrict__ positive) {
  int n = blockIdx.x * blockDim.x + threadIdx.x;
  if (n >= NROW_) return;
  int b = n / HW_, hw = n % HW_;
  const float* p1 = x1 + (size_t)b * D_ * HW_ + hw;   // stride HW_ over d
  const float* p2 = x2 + (size_t)b * D_ * HW_ + hw;

  float s1 = 0.f, s2 = 0.f;
  #pragma unroll 8
  for (int d = 0; d < D_; ++d) {
    float a = p1[d * HW_];
    float c = p2[d * HW_];
    s1 += a * a;
    s2 += c * c;
  }
  float i1 = 1.f / fmaxf(sqrtf(s1), EPS_);
  float i2 = 1.f / fmaxf(sqrtf(s2), EPS_);

  float pos = 0.f;
  _Float16* orow = x1h + (size_t)n * D_;
  for (int d8 = 0; d8 < D_ / 8; ++d8) {
    v8h pk;
    #pragma unroll
    for (int i = 0; i < 8; ++i) {
      int d = d8 * 8 + i;
      float a = p1[d * HW_] * i1;
      float c = p2[d * HW_] * i2;
      pos += __expf(a * c);          // elementwise positive similarity
      pk[i] = (_Float16)a;
    }
    *(v8h*)(orow + d8 * 8) = pk;     // 16B aligned store
  }
  positive[n] = pos;
}

// ---------------------------------------------------------------- phase 2 ----
// 16-bit A layout (16x32): lanes 0-15 = M, K blocks {0..7, 16..23};
//                          lanes 16-31 = M, K blocks {8..15, 24..31}.
// B as loaded here (32x16, same K order as the A loads): lane = N column; lanes 0-15 hold K=0..15,
//                          lanes 16-31 hold K=16..31 (2 halves per VGPR).
// D layout (16x16 f32): VGPR r: lanes 0-15 -> M=r, lanes 16-31 -> M=8+r.
__global__ void nce_negative_kernel(const int* __restrict__ neg_idx,
                                    const _Float16* __restrict__ x1h,
                                    const float* __restrict__ positive,
                                    float* __restrict__ out) {
  __shared__ float wsum[8];
  const int lane  = threadIdx.x & 31;
  const int wave  = threadIdx.x >> 5;
  const int n     = blockIdx.x * 8 + wave;       // 3136 blocks * 8 waves = 25088
  const int khalf = lane >> 4;
  const int mlo   = lane & 15;

  // B operands: row vector v replicated across all 16 columns.
  const _Float16* vrow = x1h + (size_t)n * D_;
  v16h B0 = *(const v16h*)(vrow + khalf * 16);        // K = 0..31 block
  v16h B1 = *(const v16h*)(vrow + 32 + khalf * 16);   // K = 32..63 block

  const int* idxrow = neg_idx + (size_t)n * KNEG_;
  float negsum = 0.f;

  for (int t = 0; t < 7; ++t) {
    int m  = t * 16 + mlo;
    int mc = (m < KNEG_) ? m : (KNEG_ - 1);           // clamp tail, mask later
    int j  = idxrow[mc];
    const _Float16* arow = x1h + (size_t)j * D_;      // gathered negative row

    union { v16h v; v8h h[2]; } A0, A1;
    A0.h[0] = *(const v8h*)(arow + khalf * 16);       // K 0..7   / 16..23
    A0.h[1] = *(const v8h*)(arow + 8 + khalf * 16);   // K 8..15  / 24..31
    A1.h[0] = *(const v8h*)(arow + 32 + khalf * 16);  // K 32..39 / 48..55
    A1.h[1] = *(const v8h*)(arow + 40 + khalf * 16);  // K 40..47 / 56..63

    v8f c = {};
    c = __builtin_amdgcn_wmma_f32_16x16x32_f16(false, A0.v, false, B0,
                                               (short)0, c, false, false);
    c = __builtin_amdgcn_wmma_f32_16x16x32_f16(false, A1.v, false, B1,
                                               (short)0, c, false, false);

    // c[r] = dot(neg_{mbase+r}, v) replicated across columns.
    int mbase = t * 16 + khalf * 8;
    #pragma unroll
    for (int r = 0; r < 8; ++r)
      negsum += (mbase + r < KNEG_) ? __expf(c[r]) : 0.f;
  }

  // lanes 0..15 hold sum over M=0..7 of each tile, lanes 16..31 over M=8..15
  float total = __shfl(negsum, 0) + __shfl(negsum, 16);

  if (lane == 0) {
    float p = positive[n];
    wsum[wave] = __logf(p + total) - __logf(p);       // -(log p - log(p+neg))
  }
  __syncthreads();
  if (threadIdx.x == 0) {
    float s = 0.f;
    #pragma unroll
    for (int w = 0; w < 8; ++w) s += wsum[w];
    atomicAdd(out, s * (1.0f / NROW_));
  }
}

// ----------------------------------------------------------------- launch ----
extern "C" void kernel_launch(void* const* d_in, const int* in_sizes, int n_in,
                              void* d_out, int out_size, void* d_ws, size_t ws_size,
                              hipStream_t stream) {
  const float* x1      = (const float*)d_in[0];
  const float* x2      = (const float*)d_in[1];
  const int*   neg_idx = (const int*)d_in[2];
  float*       out     = (float*)d_out;

  _Float16* x1h      = (_Float16*)d_ws;                         // 25088*64*2 B
  float*    positive = (float*)((char*)d_ws + (size_t)NROW_ * D_ * 2);

  hipMemsetAsync(d_out, 0, sizeof(float), stream);

  nce_normalize_kernel<<<(NROW_ + 255) / 256, 256, 0, stream>>>(x1, x2, x1h, positive);
  nce_negative_kernel<<<NROW_ / 8, 256, 0, stream>>>(neg_idx, x1h, positive, out);
}
